// PS_APM_Seq_6390911336480
// MI455X (gfx1250) — matrix operand order repaired and checked by reading, not compiled, not run on hardware
//
#include <hip/hip_runtime.h>
#include <cstdint>

// ---------------- problem constants ----------------
#define BB   16
#define NN   96
#define CC   4
#define TT   256
#define HH   128
#define DTt  16
#define DPp  128
#define TOUT 64
#define DINP 160   // DIN=144 padded to 5 k-tiles of 32
#define BN   (BB*NN)

typedef __attribute__((ext_vector_type(16))) _Float16 v16h;
typedef __attribute__((ext_vector_type(8)))  _Float16 v8h;
typedef __attribute__((ext_vector_type(8)))  float    v8f;
typedef __attribute__((ext_vector_type(4)))  unsigned int v4u;
typedef __attribute__((ext_vector_type(8)))  int      v8i;
typedef __attribute__((ext_vector_type(4)))  int      v4i;

__device__ __forceinline__ int lane_id() { return (int)(threadIdx.x & 31u); }

__device__ __forceinline__ v8f wmma32(v16h a, v16h b, v8f c) {
  return __builtin_amdgcn_wmma_f32_16x16x32_f16(false, a, false, b, (short)0, c, false, false);
}

// ---- Tensor Data Mover: 1-D f16 tile (n_elems < 65536) global -> LDS ----
// D# layout per CDNA5 ISA:
//  g0: [1:0]=count=1, [63:32]=lds_addr, [120:64]=global_addr, [127:126]=type=2
//  g1: [17:16]=data_size=1 (2B), [79:48]=tensor_dim0, [127:112]=tile_dim0
__device__ __forceinline__ void tdm_load_f16_1d(unsigned int lds_byte_off,
                                                const void* gptr,
                                                unsigned int n_elems) {
  unsigned long long ga = (unsigned long long)(size_t)gptr;
  v4u g0;
  g0[0] = 1u;                                                    // count=1 (user D#)
  g0[1] = lds_byte_off;                                          // lds_addr
  g0[2] = (unsigned int)(ga & 0xffffffffu);                      // global_addr[31:0]
  g0[3] = (unsigned int)((ga >> 32) & 0x01ffffffu) | 0x80000000u;// [56:32] + type=2
  v8i g1;
  g1[0] = 0x00010000;                                            // data_size=1 -> 2 bytes
  g1[1] = (int)((n_elems & 0xffffu) << 16);                      // tensor_dim0[15:0]
  g1[2] = (int)((n_elems >> 16) & 0xffffu);                      // tensor_dim0[31:16]
  g1[3] = (int)((n_elems & 0xffffu) << 16);                      // tile_dim0
  g1[4] = 0;                                                     // tile_dim1/2 unused
  g1[5] = (int)n_elems;                                          // dim0 stride (benign, 1-D)
  g1[6] = 0;
  g1[7] = 0;
  v4i z4 = {0, 0, 0, 0};
#if defined(__clang_major__) && (__clang_major__ >= 23)
  v8i z8 = {0, 0, 0, 0, 0, 0, 0, 0};
  __builtin_amdgcn_tensor_load_to_lds(g0, g1, z4, z4, z8, 0);
#else
  __builtin_amdgcn_tensor_load_to_lds(g0, g1, z4, z4, 0);
#endif
}

// A fragment (16x32, f16) from row-major f16, row stride ld (elements).
__device__ __forceinline__ v16h ldA16(const _Float16* src, int ld) {
  const int l = lane_id();
  const int m = l & 15;
  const int ko = (l >> 4) << 3;          // 0 or 8
  const _Float16* p = src + (size_t)m * ld;
  v16h a;
#pragma unroll
  for (int i = 0; i < 8; ++i) { a[i] = p[ko + i]; a[i + 8] = p[16 + ko + i]; }
  return a;
}

// Same but source is f32 (e.g. LDS-resident hidden state), converted on load.
__device__ __forceinline__ v16h ldA16_f32(const float* src, int ld) {
  const int l = lane_id();
  const int m = l & 15;
  const int ko = (l >> 4) << 3;
  const float* p = src + (size_t)m * ld;
  v16h a;
#pragma unroll
  for (int i = 0; i < 8; ++i) { a[i] = (_Float16)p[ko + i]; a[i + 8] = (_Float16)p[16 + ko + i]; }
  return a;
}

// B fragment (32x16, f16) from a weight stored (N_out, K_in) row-major:
// B[k][n] = W[n][k]; per-lane load is two runs of 8 halfs (k0..k0+7, 16+k0..16+k0+7).
__device__ __forceinline__ v16h ldB16_wt(const _Float16* w, int ld) {
  const int l = lane_id();
  const int n = l & 15;
  const int k0 = (l >> 4) << 3;          // 0 or 8
  const _Float16* p = w + (size_t)n * ld + k0;
  v16h b;
#pragma unroll
  for (int i = 0; i < 8; ++i) { b[i] = p[i]; b[i + 8] = p[16 + i]; }
  return b;
}

// B fragment from K-major storage: B[k][n] = src[k*ld + n].
__device__ __forceinline__ v16h ldB16_kn(const _Float16* src, int ld) {
  const int l = lane_id();
  const int n = l & 15;
  const int k0 = (l >> 4) << 3;
  v16h b;
#pragma unroll
  for (int i = 0; i < 8; ++i) { b[i] = src[(size_t)(k0 + i) * ld + n]; b[i + 8] = src[(size_t)(16 + k0 + i) * ld + n]; }
  return b;
}

__device__ __forceinline__ void stD_f32(float* dst, int ld, v8f d) {
  const int l = lane_id();
  const int n = l & 15;
  const int mb = (l >> 4) << 3;
#pragma unroll
  for (int r = 0; r < 8; ++r) dst[(size_t)(mb + r) * ld + n] = d[r];
}

__device__ __forceinline__ void stD_f16(_Float16* dst, int ld, v8f d) {
  const int l = lane_id();
  const int n = l & 15;
  const int mb = (l >> 4) << 3;
#pragma unroll
  for (int r = 0; r < 8; ++r) dst[(size_t)(mb + r) * ld + n] = (_Float16)d[r];
}

__device__ __forceinline__ float sigm(float x) { return 1.f / (1.f + __expf(-x)); }
__device__ __forceinline__ float softplusf(float x) { return x > 20.f ? x : log1pf(__expf(x)); }

// ---------------- prep kernels ----------------

__global__ void pad_convert(const float* __restrict__ src, _Float16* __restrict__ dst,
                            int rows, int sc, int dc) {
  int total = rows * dc;
  for (int i = blockIdx.x * blockDim.x + threadIdx.x; i < total; i += gridDim.x * blockDim.x) {
    int r = i / dc, c = i - r * dc;
    dst[i] = (_Float16)(c < sc ? src[(size_t)r * sc + c] : 0.f);
  }
}

__global__ void make_utab(const float* __restrict__ tw, const float* __restrict__ tb,
                          _Float16* __restrict__ u) {
  int i = blockIdx.x * blockDim.x + threadIdx.x;
  if (i < TT * DTt) {
    int t = i >> 4, k = i & 15;
    u[i] = (_Float16)tanhf(((float)t / 255.f) * tw[k] + tb[k]);
  }
}

__global__ __launch_bounds__(128) void make_A(const float* __restrict__ S,
                                              const float* __restrict__ G,
                                              const int* __restrict__ phases,
                                              float* __restrict__ Af,
                                              _Float16* __restrict__ A16) {
  const int b = blockIdx.x;
  const int p = phases[b];
  const int tid = threadIdx.x;
  __shared__ float red[128];
  float v = 0.f;
  if (tid < NN) v = softplusf(G[p * NN + tid]) + 1e-6f;
  red[tid] = v;
  __syncthreads();
  for (int o = 64; o >= 1; o >>= 1) { if (tid < o) red[tid] += red[tid + o]; __syncthreads(); }
  const float ggsum = fmaxf(red[0], 1e-6f);
  __syncthreads();
  if (tid < NN) {
    const int i = tid;
    const float gg = (softplusf(G[p * NN + i]) + 1e-6f) * ((float)NN / ggsum);
    const float* srow = S + ((size_t)p * NN + i) * NN;
    float denom = 0.f;
    for (int j = 0; j < NN; ++j) if (j != i) denom += fabsf(srow[j]);
    denom = fmaxf(denom, 1e-6f);
    float* arow = Af + ((size_t)b * NN + i) * NN;
    _Float16* arow16 = A16 + ((size_t)b * NN + i) * NN;
    for (int j = 0; j < NN; ++j) {
      float s0 = (j == i) ? 0.f : srow[j];
      float a = (s0 / denom) * gg;
      arow[j] = a;
      arow16[j] = (_Float16)a;
    }
  }
}

__global__ __launch_bounds__(256) void spec_norm(const float* __restrict__ W,
                                                 const float* __restrict__ u,
                                                 _Float16* __restrict__ Wsn16) {
  const int tid = threadIdx.x;
  __shared__ float sv[144];
  __shared__ float su2[128];
  __shared__ float red[256];
  for (int k = tid; k < 144; k += 256) {
    float acc = 0.f;
    for (int i = 0; i < 128; ++i) acc += W[(size_t)i * 144 + k] * u[i];
    sv[k] = acc;
  }
  __syncthreads();
  float part = 0.f;
  for (int k = tid; k < 144; k += 256) part += sv[k] * sv[k];
  red[tid] = part; __syncthreads();
  for (int o = 128; o >= 1; o >>= 1) { if (tid < o) red[tid] += red[tid + o]; __syncthreads(); }
  const float nv = sqrtf(red[0]) + 1e-12f;
  __syncthreads();
  for (int k = tid; k < 144; k += 256) sv[k] /= nv;
  __syncthreads();
  for (int i = tid; i < 128; i += 256) {
    float acc = 0.f;
    for (int k = 0; k < 144; ++k) acc += W[(size_t)i * 144 + k] * sv[k];
    su2[i] = acc;
  }
  __syncthreads();
  part = 0.f;
  for (int i = tid; i < 128; i += 256) part += su2[i] * su2[i];
  red[tid] = part; __syncthreads();
  for (int o = 128; o >= 1; o >>= 1) { if (tid < o) red[tid] += red[tid + o]; __syncthreads(); }
  const float nu = sqrtf(red[0]) + 1e-12f;
  __syncthreads();
  part = 0.f;
  for (int i = tid; i < 128; i += 256) part += (su2[i] / nu) * su2[i];  // sigma = u2n . (W@v)
  red[tid] = part; __syncthreads();
  for (int o = 128; o >= 1; o >>= 1) { if (tid < o) red[tid] += red[tid + o]; __syncthreads(); }
  const float sigma = red[0];
  __syncthreads();
  for (int i = tid; i < 128 * DINP; i += 256) {
    int r = i / DINP, c = i - r * DINP;
    Wsn16[i] = (_Float16)(c < 144 ? W[(size_t)r * 144 + c] / sigma : 0.f);
  }
}

// ---------------- stage 1: per-node GRU encoder + LayerNorm ----------------
// one workgroup per node; Whh (96KB f16) staged once via TDM into LDS and
// reused for all 256 steps; dynamic LDS = 152KB (CDNA5 WGP has 320KB).
#define ENC_LDS (16*128*4 + 16*384*4 + 16*384*4 + 384*128*2)

__global__ __launch_bounds__(256) void enc_gru(const float* __restrict__ X,
                                               const float* __restrict__ Wih,
                                               const _Float16* __restrict__ Whh16,
                                               const float* __restrict__ bih,
                                               const float* __restrict__ bhh,
                                               const float* __restrict__ lng,
                                               const float* __restrict__ lnb,
                                               const _Float16* __restrict__ utab,
                                               _Float16* __restrict__ Hc) {
  const int n = blockIdx.x;
  const int tid = threadIdx.x;
  const int wave = tid >> 5;
  extern __shared__ char esmem[];
  float*    sh  = (float*)esmem;              // 16*128
  float*    sgi = sh + 16 * 128;              // 16*384
  float*    sgh = sgi + 16 * 384;             // 16*384
  _Float16* sW  = (_Float16*)(sgh + 16 * 384);// 384*128 Whh tile

  const _Float16* Whh_n = Whh16 + (size_t)n * 384 * 128;
  // TDM: DMA this node's recurrent weights into LDS (one descriptor, wave 0)
  if (wave == 0) {
    tdm_load_f16_1d((unsigned int)(size_t)(void*)sW, (const void*)Whh_n, 384 * 128);
    __builtin_amdgcn_s_wait_tensorcnt(0);
  }
  for (int i = tid; i < 16 * 128; i += 256) sh[i] = 0.f;
  __syncthreads();

  const float* Wih_n = Wih + (size_t)n * 384 * 4;
  const float* bih_n = bih + (size_t)n * 384;
  const float* bhh_n = bhh + (size_t)n * 384;

  for (int t = 0; t < TT; ++t) {
    // gi = x_t @ Wih^T + bih  (K=4: VALU)
    for (int e = tid; e < 16 * 384; e += 256) {
      int b = e / 384, g = e - b * 384;
      const float* xp = X + ((size_t)(b * NN + n) * CC) * TT + t;
      const float* w = Wih_n + (size_t)g * 4;
      float acc = bih_n[g];
#pragma unroll
      for (int c = 0; c < CC; ++c) acc += xp[(size_t)c * TT] * w[c];
      sgi[e] = acc;
    }
    // gh = h @ Whh^T + bhh  (WMMA: M=16 batch, K=128, N=384; B from LDS)
    {
      v16h hA[4];
#pragma unroll
      for (int kt = 0; kt < 4; ++kt) hA[kt] = ldA16_f32(sh + kt * 32, 128);
      for (int jt = wave * 3; jt < wave * 3 + 3; ++jt) {
        const int g0 = jt * 16;
        const float bv = bhh_n[g0 + (tid & 15)];
        v8f acc;
#pragma unroll
        for (int i = 0; i < 8; ++i) acc[i] = bv;
#pragma unroll
        for (int kt = 0; kt < 4; ++kt)
          acc = wmma32(hA[kt], ldB16_wt(sW + (size_t)g0 * 128 + kt * 32, 128), acc);
        stD_f32(sgh + g0, 384, acc);
      }
    }
    __syncthreads();
    // gates + state update
    for (int e = tid; e < 16 * 128; e += 256) {
      int b = e >> 7, j = e & 127;
      const float* gi = sgi + (size_t)b * 384;
      const float* gh = sgh + (size_t)b * 384;
      float r = sigm(gi[j] + gh[j]);
      float z = sigm(gi[128 + j] + gh[128 + j]);
      float nn2 = tanhf(gi[256 + j] + r * gh[256 + j]);
      sh[e] = (1.f - z) * nn2 + z * sh[e];
    }
    __syncthreads();
    // LayerNorm over H=128 per batch + emit Hc row (f16, padded to 160)
    {
      const int b = tid >> 4, s = tid & 15;
      const float* hb = sh + (size_t)b * 128;
      float lsum = 0.f, lsq = 0.f;
#pragma unroll
      for (int i = 0; i < 8; ++i) { float v = hb[s * 8 + i]; lsum += v; lsq += v * v; }
#pragma unroll
      for (int m = 8; m >= 1; m >>= 1) { lsum += __shfl_xor(lsum, m); lsq += __shfl_xor(lsq, m); }
      const float mu = lsum * (1.f / 128.f);
      const float var = lsq * (1.f / 128.f) - mu * mu;
      const float rstd = rsqrtf(var + 1e-5f);
      _Float16* out = Hc + (((size_t)(b * NN + n)) * TT + t) * DINP;
#pragma unroll
      for (int i = 0; i < 8; ++i) {
        int j = s * 8 + i;
        out[j] = (_Float16)((hb[j] - mu) * rstd * lng[j] + lnb[j]);
      }
      out[128 + s] = utab[t * DTt + s];
      out[144 + s] = (_Float16)0.f;
    }
    __syncthreads();
  }
}

// ---------------- stage 2a: Hs = Hc@Wself^T, Hn = Hc@Wsn^T ----------------
__global__ __launch_bounds__(256) void proj_hs_hn(const _Float16* __restrict__ Hc,
                                                  const _Float16* __restrict__ Wself16,
                                                  const _Float16* __restrict__ Wsn16,
                                                  _Float16* __restrict__ Hs,
                                                  _Float16* __restrict__ Hn) {
  const int wave = threadIdx.x >> 5;
  const int mt = blockIdx.x * 8 + wave;
  const int r0 = mt * 16;                 // row = bn*256 + t (16 rows share bn)
  const int bn = r0 >> 8, t0 = r0 & 255;
  const int b = bn / NN, nd = bn - b * NN;
  const _Float16* arow = Hc + (size_t)r0 * DINP;
  v16h A[5];
#pragma unroll
  for (int k = 0; k < 5; ++k) A[k] = ldA16(arow + k * 32, DINP);
  const size_t obase = (((size_t)b * TT + t0) * NN + nd) * DPp;
#pragma unroll
  for (int w = 0; w < 2; ++w) {
    const _Float16* W = w ? Wsn16 : Wself16;
    _Float16* O = (w ? Hn : Hs) + obase;
#pragma unroll
    for (int nt = 0; nt < 8; ++nt) {
      v8f acc = {};
#pragma unroll
      for (int k = 0; k < 5; ++k)
        acc = wmma32(A[k], ldB16_wt(W + (size_t)(nt * 16) * DINP + k * 32, DINP), acc);
      stD_f16(O + nt * 16, NN * DPp, acc);     // row stride = one t step
    }
  }
}

// ---------------- stage 2b: Z = leaky(Hs + A@Hn), per (b,t) tile ----------------
__global__ __launch_bounds__(256) void mix_zn(const _Float16* __restrict__ A16,
                                              const _Float16* __restrict__ Hn,
                                              const _Float16* __restrict__ Hs,
                                              _Float16* __restrict__ Z) {
  const int t = blockIdx.x, b = blockIdx.y;
  __shared__ _Float16 sHn[NN * DPp];      // 24KB, filled by TDM
  const size_t base = (((size_t)b * TT + t) * NN) * DPp;
  if ((threadIdx.x >> 5) == 0) {
    tdm_load_f16_1d((unsigned int)(size_t)(void*)sHn, (const void*)(Hn + base), NN * DPp);
    __builtin_amdgcn_s_wait_tensorcnt(0);
  }
  __syncthreads();
  const int wave = threadIdx.x >> 5;
  const _Float16* Ab = A16 + (size_t)b * NN * NN;
  for (int q = 0; q < 6; ++q) {
    const int tile = wave * 6 + q;        // 48 tiles: 6 node-tiles x 8 d-tiles
    const int it = tile >> 3, dt = tile & 7;
    v8f acc = {};
#pragma unroll
    for (int k = 0; k < 3; ++k) {
      v16h a = ldA16(Ab + (size_t)(it * 16) * NN + k * 32, NN);
      v16h bb = ldB16_kn(sHn + (size_t)(k * 32) * DPp + dt * 16, DPp);
      acc = wmma32(a, bb, acc);
    }
    const int l = lane_id();
    const int nc = l & 15, mb = (l >> 4) << 3;
#pragma unroll
    for (int r = 0; r < 8; ++r) {
      const int i = it * 16 + mb + r;     // node
      const int d = dt * 16 + nc;
      float v = acc[r] + (float)Hs[base + (size_t)i * DPp + d];
      v = v >= 0.f ? v : 0.1f * v;
      Z[(((size_t)(b * NN + i)) * TT + t) * DPp + d] = (_Float16)v;
    }
  }
}

// ---------------- stage 2c: K = Z@k_w^T, V = Z@v_w^T ----------------
__global__ __launch_bounds__(256) void kv_proj(const _Float16* __restrict__ Z,
                                               const _Float16* __restrict__ kw16,
                                               const _Float16* __restrict__ vw16,
                                               _Float16* __restrict__ Kf,
                                               _Float16* __restrict__ Vf) {
  const int wave = threadIdx.x >> 5;
  const int mt = blockIdx.x * 8 + wave;
  const size_t r0 = (size_t)mt * 16;
  const _Float16* arow = Z + r0 * DPp;
  v16h A[4];
#pragma unroll
  for (int k = 0; k < 4; ++k) A[k] = ldA16(arow + k * 32, DPp);
#pragma unroll
  for (int w = 0; w < 2; ++w) {
    const _Float16* W = w ? vw16 : kw16;
    _Float16* O = (w ? Vf : Kf) + r0 * DPp;
#pragma unroll
    for (int nt = 0; nt < 8; ++nt) {
      v8f acc = {};
#pragma unroll
      for (int k = 0; k < 4; ++k)
        acc = wmma32(A[k], ldB16_wt(W + (size_t)(nt * 16) * DPp + k * 32, DPp), acc);
      stD_f16(O + nt * 16, DPp, acc);
    }
  }
}

// ---------------- stage 3: attention + GRU decode head ----------------
#define HEAD_LDS (NN*NN*4 + NN*DPp*4 + NN*CC*4 + NN*CC*4 + NN*TT*4 + NN*DINP*2 + NN*DPp*2)

__global__ __launch_bounds__(512) void head_kernel(const _Float16* __restrict__ Kf,
                                                   const _Float16* __restrict__ Vf,
                                                   const _Float16* __restrict__ Z,
                                                   const float* __restrict__ Af,
                                                   const _Float16* __restrict__ cWih16,
                                                   const _Float16* __restrict__ cWhh16,
                                                   const float* __restrict__ cbih,
                                                   const float* __restrict__ cbhh,
                                                   const float* __restrict__ rw,
                                                   const float* __restrict__ rb,
                                                   const float* __restrict__ ltau,
                                                   const float* __restrict__ X,
                                                   float* __restrict__ out) {
  const int b = blockIdx.x;
  const int tid = threadIdx.x;
  const int wave = tid >> 5;
  extern __shared__ char smem[];
  float*    sA   = (float*)smem;                  // 96*96
  float*    sh   = sA + NN * NN;                  // 96*128
  float*    sy   = sh + NN * DPp;                 // 96*4
  float*    sy2  = sy + NN * CC;                  // 96*4
  float*    sS   = sy2 + NN * CC;                 // 96*256 scores/alpha
  _Float16* sInp = (_Float16*)(sS + NN * TT);     // 96*160
  _Float16* sH16 = sInp + NN * DINP;              // 96*128

  const float scale = __expf(ltau[0]);

  for (int i = tid; i < NN * NN; i += 512) sA[i] = Af[(size_t)b * NN * NN + i];
  for (int i = tid; i < NN * DPp; i += 512) {
    const int m = i >> 7;
    sh[i] = (float)Z[(((size_t)(b * NN + m)) * TT + (TT - 1)) * DPp + (i & 127)];
  }
  for (int i = tid; i < NN * CC; i += 512) sy[i] = 0.f;
  for (int i = tid; i < NN * DINP; i += 512) sInp[i] = (_Float16)0.f;
  __syncthreads();

  for (int s = 0; s < TOUT; ++s) {
    // (1) y propagation through adjacency
    for (int e = tid; e < NN * CC; e += 512) {
      const int i = e >> 2, c = e & 3;
      const float* ar = sA + (size_t)i * NN;
      float acc = 0.f;
      for (int j = 0; j < NN; ++j) acc += ar[j] * sy[j * CC + c];
      sy2[e] = sy[e] + 0.3f * acc;
    }
    __syncthreads();
    // (2) scores[m][t] = scale * K[m,t,:].h[m,:]
    for (int e = tid; e < NN * TT; e += 512) {
      const int m = e >> 8, t = e & 255;
      const _Float16* kp = Kf + (((size_t)(b * NN + m)) * TT + t) * DPp;
      const float* hp = sh + (size_t)m * DPp;
      float acc = 0.f;
#pragma unroll
      for (int d = 0; d < DPp; d += 8) {
        v8h kv = *(const v8h*)(kp + d);
#pragma unroll
        for (int i2 = 0; i2 < 8; ++i2) acc += (float)kv[i2] * hp[d + i2];
      }
      sS[e] = acc * scale;
    }
    __syncthreads();
    // (3) softmax over T per row (6 rows per wave)
    for (int rr = 0; rr < 6; ++rr) {
      const int m = wave * 6 + rr;
      const int l = tid & 31;
      float mx = -1e30f;
      for (int t = l; t < TT; t += 32) mx = fmaxf(mx, sS[m * TT + t]);
#pragma unroll
      for (int o = 16; o >= 1; o >>= 1) mx = fmaxf(mx, __shfl_xor(mx, o));
      float sum = 0.f;
      for (int t = l; t < TT; t += 32) { float e2 = __expf(sS[m * TT + t] - mx); sS[m * TT + t] = e2; sum += e2; }
#pragma unroll
      for (int o = 16; o >= 1; o >>= 1) sum += __shfl_xor(sum, o);
      const float inv = 1.f / sum;
      for (int t = l; t < TT; t += 32) sS[m * TT + t] *= inv;
    }
    __syncthreads();
    // (4) ctx -> sInp[:,0:128]; y -> sInp[:,128:132]; stale h copy to f16
    for (int e = tid; e < NN * DPp; e += 512) {
      const int m = e >> 7, d = e & 127;
      const _Float16* vp = Vf + (((size_t)(b * NN + m)) * TT) * DPp + d;
      const float* al = sS + (size_t)m * TT;
      float acc = 0.f;
      for (int t = 0; t < TT; ++t) {
        if ((t & 31) == 0) __builtin_prefetch(vp + (size_t)(t + 64) * DPp, 0, 1);
        acc += al[t] * (float)vp[(size_t)t * DPp];
      }
      sInp[m * DINP + d] = (_Float16)acc;
      sH16[e] = (_Float16)sh[e];
    }
    for (int e = tid; e < NN * CC; e += 512) sInp[(e >> 2) * DINP + 128 + (e & 3)] = (_Float16)sy2[e];
    __syncthreads();
    // (5) GRU cell via WMMA: 48 (m-tile, gate-col-tile) tasks, 3 per wave.
    for (int q = 0; q < 3; ++q) {
      const int task = wave * 3 + q;
      const int it = task >> 3, jt = task & 7;
      v16h Ai[5];
#pragma unroll
      for (int k = 0; k < 5; ++k) Ai[k] = ldA16(sInp + (size_t)(it * 16) * DINP + k * 32, DINP);
      v16h Ah[4];
#pragma unroll
      for (int k = 0; k < 4; ++k) Ah[k] = ldA16(sH16 + (size_t)(it * 16) * DPp + k * 32, DPp);
      const int l = lane_id();
      const int nc = l & 15, mb = (l >> 4) << 3;
      const int cr = jt * 16 + nc;
      v8f gir, giz, gin, ghr, ghz, ghn;
      {
        const float b0 = cbih[cr], b1 = cbih[128 + cr], b2 = cbih[256 + cr];
        const float c0 = cbhh[cr], c1 = cbhh[128 + cr], c2 = cbhh[256 + cr];
#pragma unroll
        for (int i = 0; i < 8; ++i) { gir[i] = b0; giz[i] = b1; gin[i] = b2; ghr[i] = c0; ghz[i] = c1; ghn[i] = c2; }
      }
#pragma unroll
      for (int k = 0; k < 5; ++k) {
        gir = wmma32(Ai[k], ldB16_wt(cWih16 + (size_t)(jt * 16) * DINP + k * 32, DINP), gir);
        giz = wmma32(Ai[k], ldB16_wt(cWih16 + (size_t)(128 + jt * 16) * DINP + k * 32, DINP), giz);
        gin = wmma32(Ai[k], ldB16_wt(cWih16 + (size_t)(256 + jt * 16) * DINP + k * 32, DINP), gin);
      }
#pragma unroll
      for (int k = 0; k < 4; ++k) {
        ghr = wmma32(Ah[k], ldB16_wt(cWhh16 + (size_t)(jt * 16) * DPp + k * 32, DPp), ghr);
        ghz = wmma32(Ah[k], ldB16_wt(cWhh16 + (size_t)(128 + jt * 16) * DPp + k * 32, DPp), ghz);
        ghn = wmma32(Ah[k], ldB16_wt(cWhh16 + (size_t)(256 + jt * 16) * DPp + k * 32, DPp), ghn);
      }
      const int jcol = jt * 16 + nc;
#pragma unroll
      for (int r = 0; r < 8; ++r) {
        const int m = it * 16 + mb + r;
        const float rg = sigm(gir[r] + ghr[r]);
        const float zg = sigm(giz[r] + ghz[r]);
        const float ng = tanhf(gin[r] + rg * ghn[r]);
        const float hv = sh[(size_t)m * DPp + jcol];
        sh[(size_t)m * DPp + jcol] = (1.f - zg) * ng + zg * hv;     // exclusive tile
      }
    }
    __syncthreads();
    // (6) y_t = h@rw^T + rb + y ; emit output
    for (int e = tid; e < NN * CC; e += 512) {
      const int m = e >> 2, c = e & 3;
      const float* hp = sh + (size_t)m * DPp;
      const float* wp = rw + (size_t)c * DPp;
      float acc = rb[c] + sy2[e];
      for (int d = 0; d < DPp; ++d) acc += hp[d] * wp[d];
      sy[e] = acc;
      const int gm = b * NN + m;
      const float xl = X[(((size_t)gm) * CC + c) * TT + (TT - 1)];
      out[((size_t)gm * CC + c) * TOUT + s] = xl + acc;
      if (s == 0) out[(size_t)BN * CC * TOUT + (size_t)gm * CC + c] = xl + acc;
    }
    __syncthreads();
  }
}

// ---------------- host launcher ----------------
extern "C" void kernel_launch(void* const* d_in, const int* in_sizes, int n_in,
                              void* d_out, int out_size, void* d_ws, size_t ws_size,
                              hipStream_t stream) {
  (void)in_sizes; (void)n_in; (void)out_size; (void)ws_size;
  const float* X      = (const float*)d_in[0];
  const int*   phases = (const int*)d_in[1];
  const float* gWih   = (const float*)d_in[2];
  const float* gWhh   = (const float*)d_in[3];
  const float* gbih   = (const float*)d_in[4];
  const float* gbhh   = (const float*)d_in[5];
  const float* lng    = (const float*)d_in[6];
  const float* lnb    = (const float*)d_in[7];
  const float* tposw  = (const float*)d_in[8];
  const float* tposb  = (const float*)d_in[9];
  const float* S      = (const float*)d_in[10];
  const float* G      = (const float*)d_in[11];
  const float* Wself  = (const float*)d_in[12];
  const float* Wneigh = (const float*)d_in[13];
  const float* snu    = (const float*)d_in[14];
  const float* kw     = (const float*)d_in[15];
  const float* vw     = (const float*)d_in[16];
  const float* cWih   = (const float*)d_in[17];
  const float* cWhh   = (const float*)d_in[18];
  const float* cbih   = (const float*)d_in[19];
  const float* cbhh   = (const float*)d_in[20];
  const float* rw     = (const float*)d_in[21];
  const float* rb     = (const float*)d_in[22];
  const float* ltau   = (const float*)d_in[23];
  float* out = (float*)d_out;

  char* ws = (char*)d_ws;
  size_t off = 0;
  auto take = [&](size_t bytes) -> char* {
    char* p = ws + off;
    off = (off + bytes + 255) & ~(size_t)255;
    return p;
  };
  _Float16* utab    = (_Float16*)take((size_t)TT * DTt * 2);
  float*    Af      = (float*)   take((size_t)BB * NN * NN * 4);
  _Float16* A16     = (_Float16*)take((size_t)BB * NN * NN * 2);
  _Float16* Wself16 = (_Float16*)take((size_t)128 * DINP * 2);
  _Float16* Wsn16   = (_Float16*)take((size_t)128 * DINP * 2);
  _Float16* kw16    = (_Float16*)take((size_t)128 * 128 * 2);
  _Float16* vw16    = (_Float16*)take((size_t)128 * 128 * 2);
  _Float16* cWih16  = (_Float16*)take((size_t)384 * DINP * 2);
  _Float16* cWhh16  = (_Float16*)take((size_t)384 * 128 * 2);
  _Float16* Whh16   = (_Float16*)take((size_t)NN * 384 * 128 * 2);
  _Float16* Hc      = (_Float16*)take((size_t)BN * TT * DINP * 2);     // aliased by Z
  _Float16* Hn      = (_Float16*)take((size_t)BB * TT * NN * DPp * 2); // aliased by K
  _Float16* Hs      = (_Float16*)take((size_t)BB * TT * NN * DPp * 2); // aliased by V
  _Float16* Z  = Hc;   // Hc dead after proj
  _Float16* Kf = Hn;   // Hn dead after mix
  _Float16* Vf = Hs;   // Hs dead after mix

  pad_convert<<<2048, 256, 0, stream>>>(gWhh, Whh16, NN * 384, 128, 128);
  pad_convert<<<64, 256, 0, stream>>>(Wself, Wself16, 128, 144, DINP);
  pad_convert<<<64, 256, 0, stream>>>(kw, kw16, 128, 128, 128);
  pad_convert<<<64, 256, 0, stream>>>(vw, vw16, 128, 128, 128);
  pad_convert<<<64, 256, 0, stream>>>(cWih, cWih16, 384, 132, DINP);
  pad_convert<<<64, 256, 0, stream>>>(cWhh, cWhh16, 384, 128, 128);
  make_utab<<<16, 256, 0, stream>>>(tposw, tposb, utab);
  make_A<<<BB, 128, 0, stream>>>(S, G, phases, Af, A16);
  spec_norm<<<1, 256, 0, stream>>>(Wneigh, snu, Wsn16);

  (void)hipFuncSetAttribute((const void*)enc_gru,
                            hipFuncAttributeMaxDynamicSharedMemorySize, (int)ENC_LDS);
  enc_gru<<<NN, 256, ENC_LDS, stream>>>(X, gWih, Whh16, gbih, gbhh, lng, lnb, utab, Hc);
  proj_hs_hn<<<(BN * TT / 16) / 8, 256, 0, stream>>>(Hc, Wself16, Wsn16, Hs, Hn);
  mix_zn<<<dim3(TT, BB), 256, 0, stream>>>(A16, Hn, Hs, Z);
  kv_proj<<<(BN * TT / 16) / 8, 256, 0, stream>>>(Z, kw16, vw16, Kf, Vf);

  (void)hipFuncSetAttribute((const void*)head_kernel,
                            hipFuncAttributeMaxDynamicSharedMemorySize, (int)HEAD_LDS);
  head_kernel<<<BB, 512, HEAD_LDS, stream>>>(Kf, Vf, Z, Af, cWih16, cWhh16,
                                             cbih, cbhh, rw, rb, ltau, X, out);
}
